// MultiHeadAttention_25116968747290
// MI455X (gfx1250) — matrix operand order repaired and checked by reading, not compiled, not run on hardware
//
#include <hip/hip_runtime.h>
#include <hip/hip_bf16.h>

constexpr int   kSeq   = 2048;
constexpr int   kBatch = 2;
constexpr int   kTok   = kBatch * kSeq;   // 4096
constexpr int   kDim   = 768;
constexpr int   kHeads = 12;
constexpr int   kHd    = 64;
constexpr int   kQkv   = 3 * kDim;        // 2304

typedef __attribute__((ext_vector_type(16))) _Float16 v16h;
typedef __attribute__((ext_vector_type(8)))  float    v8f;
typedef int v4i __attribute__((vector_size(16)));

#define AS1 __attribute__((address_space(1)))
#define AS3 __attribute__((address_space(3)))

// Async global->LDS staging (gfx1250 GLOBAL_LOAD_ASYNC_TO_LDS_B128, ASYNCcnt),
// guarded so we fall back to plain load/ds_store if the builtin is absent.
// Probe-confirmed signature: (v4i AS1* gsrc, v4i AS3* ldst, imm offset, imm cpol)
#if defined(__HIP_DEVICE_COMPILE__) &&                                   \
    __has_builtin(__builtin_amdgcn_global_load_async_to_lds_b128) &&     \
    __has_builtin(__builtin_amdgcn_s_wait_asynccnt)
#define USE_ASYNC_LDS 1
#else
#define USE_ASYNC_LDS 0
#endif

#if USE_ASYNC_LDS
__device__ __forceinline__ void async_copy16(const _Float16* g, _Float16* l) {
  __builtin_amdgcn_global_load_async_to_lds_b128((AS1 v4i*)g, (AS3 v4i*)l, 0, 0);
}
#endif

__device__ __forceinline__ v8f wmma16(v16h a, v16h b, v8f c) {
  // D = A(16x32 f16) * B(32x16 f16) + C(16x16 f32)
  return __builtin_amdgcn_wmma_f32_16x16x32_f16(false, a, false, b, (short)0, c,
                                                false, false);
}

// A fragment: 16x32 (MxK) tile from row-major source, leading dim lda (halves).
__device__ __forceinline__ v16h load_frag_a(const _Float16* src, int lda) {
  const int lane = threadIdx.x & 31;
  const int m  = lane & 15;
  const int kb = (lane >> 4) << 4;   // 0 or 16
  const _Float16* p = src + m * lda + kb;
  v16h a;
#pragma unroll
  for (int g = 0; g < 2; ++g)
#pragma unroll
    for (int j = 0; j < 8; ++j)
      a[g * 8 + j] = p[g * 8 + j];
  return a;
}

// B fragment from an N-major (transposed) source: physical row = N, col = K.
// Lane = column N; lanes 0-15 hold K 0..15, lanes 16-31 hold K 16..31, so the
// per-lane 16 halves are contiguous -> two ds_load_b128.
__device__ __forceinline__ v16h load_frag_bt(const _Float16* src, int lds) {
  const int lane = threadIdx.x & 31;
  const int n  = lane & 15;
  const int kb = (lane >> 4) << 4;
  const _Float16* p = src + n * lds + kb;
  v16h b;
#pragma unroll
  for (int i = 0; i < 16; ++i) b[i] = p[i];
  return b;
}

// Transpose+convert fp32 [K][N] -> f16 [N][K].
__global__ __launch_bounds__(256)
void cvt_transpose_f16(const float* __restrict__ src, _Float16* __restrict__ dst,
                       int K, int N) {
  __shared__ float tile[32][33];
  const int k0 = blockIdx.y * 32, n0 = blockIdx.x * 32;
  const int tx = threadIdx.x & 31, ty = threadIdx.x >> 5;
#pragma unroll
  for (int i = ty; i < 32; i += 8)
    tile[i][tx] = src[(size_t)(k0 + i) * N + n0 + tx];
  __syncthreads();
#pragma unroll
  for (int i = ty; i < 32; i += 8)
    dst[(size_t)(n0 + i) * K + k0 + tx] = (_Float16)tile[tx][i];
}

__global__ __launch_bounds__(256)
void layernorm_to_f16(const float* __restrict__ x, const float* __restrict__ gamma,
                      const float* __restrict__ beta, _Float16* __restrict__ xn) {
  const int row = blockIdx.x;
  const int t = threadIdx.x;
  const float* xr = x + (size_t)row * kDim;
  __shared__ float red[256];
  const float v0 = xr[t], v1 = xr[t + 256], v2 = xr[t + 512];
  red[t] = v0 + v1 + v2;
  __syncthreads();
#pragma unroll
  for (int o = 128; o > 0; o >>= 1) {
    if (t < o) red[t] += red[t + o];
    __syncthreads();
  }
  const float mu = red[0] * (1.0f / kDim);
  __syncthreads();
  const float d0 = v0 - mu, d1 = v1 - mu, d2 = v2 - mu;
  red[t] = d0 * d0 + d1 * d1 + d2 * d2;
  __syncthreads();
#pragma unroll
  for (int o = 128; o > 0; o >>= 1) {
    if (t < o) red[t] += red[t + o];
    __syncthreads();
  }
  const float rstd = rsqrtf(red[0] * (1.0f / kDim) + 1e-5f);
  _Float16* out = xn + (size_t)row * kDim;
  out[t]       = (_Float16)(d0 * rstd * gamma[t]       + beta[t]);
  out[t + 256] = (_Float16)(d1 * rstd * gamma[t + 256] + beta[t + 256]);
  out[t + 512] = (_Float16)(d2 * rstd * gamma[t + 512] + beta[t + 512]);
}

// C[MxN] = A[MxK] * Bt[NxK]^T + bias. f16 inputs, fp32 accumulate.
// 256 threads = 8 waves; block tile 64(M) x 128(N); wave tile 32x32.
// Double-buffered K-loop: async-stage tile i+1 while computing tile i.
template <bool OUT_F16>
__global__ __launch_bounds__(256)
void gemm_wmma(const _Float16* __restrict__ A, const _Float16* __restrict__ Bt,
               const float* __restrict__ bias, void* __restrict__ Cout,
               int M, int N, int K) {
  __shared__ _Float16 As[2][64][40];    // 64x32 k-tile (+8 pad)
  __shared__ _Float16 Bs[2][128][40];   // 128(N) x 32(K) k-tile (+8 pad)
  const int tid = threadIdx.x;
  const int w = tid >> 5;
  const int wr = (w >> 2) << 5;          // 0 / 32
  const int wc = (w & 3) << 5;           // 0..96
  const int rowBase = blockIdx.y << 6;   // *64
  const int colBase = blockIdx.x << 7;   // *128
  v8f acc00 = {}, acc01 = {}, acc10 = {}, acc11 = {};

  const int ar = tid >> 2, ac = (tid & 3) << 3;   // A: 8 halves / thread
  const int br = tid >> 1, bc = (tid & 1) << 4;   // Bt: 16 halves / thread

  auto stage = [&](int buf, int k0) {
    const _Float16* asrc = A  + (size_t)(rowBase + ar) * K + k0 + ac;
    const _Float16* bsrc = Bt + (size_t)(colBase + br) * K + k0 + bc;
#if USE_ASYNC_LDS
    async_copy16(asrc, &As[buf][ar][ac]);
    async_copy16(bsrc, &Bs[buf][br][bc]);
    async_copy16(bsrc + 8, &Bs[buf][br][bc + 8]);
#else
#pragma unroll
    for (int j = 0; j < 8; ++j) As[buf][ar][ac + j] = asrc[j];
#pragma unroll
    for (int j = 0; j < 16; ++j) Bs[buf][br][bc + j] = bsrc[j];
#endif
  };

  stage(0, 0);
#if USE_ASYNC_LDS
  __builtin_amdgcn_s_wait_asynccnt(0);
#endif
  __syncthreads();

  const int nk = K >> 5;
  for (int i = 0; i < nk; ++i) {
    const int cur = i & 1;
    if (i + 1 < nk) stage(cur ^ 1, (i + 1) << 5);   // prefetch next k-tile
    const v16h a0 = load_frag_a(&As[cur][wr][0], 40);
    const v16h a1 = load_frag_a(&As[cur][wr + 16][0], 40);
    const v16h b0 = load_frag_bt(&Bs[cur][wc][0], 40);
    const v16h b1 = load_frag_bt(&Bs[cur][wc + 16][0], 40);
    acc00 = wmma16(a0, b0, acc00);
    acc01 = wmma16(a0, b1, acc01);
    acc10 = wmma16(a1, b0, acc10);
    acc11 = wmma16(a1, b1, acc11);
#if USE_ASYNC_LDS
    __builtin_amdgcn_s_wait_asynccnt(0);
#endif
    __syncthreads();
  }

  const int lane = tid & 31;
  const int hi = (lane >> 4) << 3;   // C layout: VGPR r -> row r (+8 upper half)
  const int n = lane & 15;
#pragma unroll
  for (int i = 0; i < 2; ++i)
#pragma unroll
    for (int j = 0; j < 2; ++j) {
      const v8f c = (i == 0) ? (j == 0 ? acc00 : acc01)
                             : (j == 0 ? acc10 : acc11);
      const int row0 = rowBase + wr + i * 16 + hi;
      const int col = colBase + wc + j * 16 + n;
      const float bv = bias[col];
#pragma unroll
      for (int r = 0; r < 8; ++r) {
        const float vv = c[r] + bv;
        if (OUT_F16)
          ((_Float16*)Cout)[(size_t)(row0 + r) * N + col] = (_Float16)vv;
        else
          ((float*)Cout)[(size_t)(row0 + r) * N + col] = vv;
      }
    }
}

// Flash-style attention. 128 threads (4 waves); block = 64 query rows of one
// (batch, head); each wave owns 16 query rows. Scale folded into Q fragments.
__global__ __launch_bounds__(128)
void attention_wmma(const _Float16* __restrict__ qkv,
                    _Float16* __restrict__ attn_out) {
  const int qblk = blockIdx.x;   // 0..31
  const int bh = blockIdx.y;     // 0..23
  const int b = bh / kHeads;
  const int h = bh % kHeads;
  const int tid = threadIdx.x;
  const int w = tid >> 5;
  const int lane = tid & 31;
  const int hi = (lane >> 4) << 3;
  const int n = lane & 15;

  __shared__ _Float16 Ks[32][72];     // [key][dim], contiguous for S fragments
  __shared__ _Float16 Vst[64][40];    // [dim][key] transposed for PV fragments
  __shared__ _Float16 Ps[4][16][40];  // per-wave P tile (C->A layout bounce)

  const int q0 = qblk * 64 + w * 16;
  const size_t qrow = (size_t)(b * kSeq + q0) * kQkv + (size_t)h * kHd;
  v16h aq0 = load_frag_a(qkv + qrow, kQkv);        // Q dims 0..31
  v16h aq1 = load_frag_a(qkv + qrow + 32, kQkv);   // Q dims 32..63
#pragma unroll
  for (int i = 0; i < 16; ++i) {     // fold softmax scale (0.125 exact in f16)
    aq0[i] = aq0[i] * (_Float16)0.125f;
    aq1[i] = aq1[i] * (_Float16)0.125f;
  }

  float m_run[8], l_run[8];
  v8f o0 = {}, o1 = {}, o2 = {}, o3 = {};
#pragma unroll
  for (int r = 0; r < 8; ++r) { m_run[r] = -1e30f; l_run[r] = 0.0f; }

  const int key_r = tid >> 2;        // 0..31
  const int key_c = (tid & 3) << 4;  // 0,16,32,48

  for (int jt = 0; jt < kSeq / 32; ++jt) {
    const size_t krow =
        (size_t)(b * kSeq + jt * 32 + key_r) * kQkv + (size_t)h * kHd + key_c;
    const _Float16* kp = qkv + kDim + krow;       // K block
    const _Float16* vp = qkv + 2 * kDim + krow;   // V block
#if USE_ASYNC_LDS
    async_copy16(kp, &Ks[key_r][key_c]);
    async_copy16(kp + 8, &Ks[key_r][key_c + 8]);
#else
#pragma unroll
    for (int j = 0; j < 16; ++j) Ks[key_r][key_c + j] = kp[j];
#endif
#pragma unroll
    for (int j = 0; j < 16; ++j) Vst[key_c + j][key_r] = vp[j];  // transpose
    if (jt + 1 < kSeq / 32) __builtin_prefetch(kp + (size_t)32 * kQkv, 0, 0);
#if USE_ASYNC_LDS
    __builtin_amdgcn_s_wait_asynccnt(0);
#endif
    __syncthreads();

    // S = (Q*scale) * K^T : 16 queries x 32 keys, contraction over d=64
    v8f s0 = {}, s1 = {};
#pragma unroll
    for (int kk = 0; kk < 2; ++kk) {
      const v16h a = kk ? aq1 : aq0;
      const v16h bk0 = load_frag_bt(&Ks[0][kk * 32], 72);    // keys 0..15
      const v16h bk1 = load_frag_bt(&Ks[16][kk * 32], 72);   // keys 16..31
      s0 = wmma16(a, bk0, s0);
      s1 = wmma16(a, bk1, s1);
    }

    float alpha[8];
#pragma unroll
    for (int r = 0; r < 8; ++r) {
      const float x0 = s0[r];
      const float x1 = s1[r];
      float rm = fmaxf(x0, x1);            // row reduce across 16 key lanes
      rm = fmaxf(rm, __shfl_xor(rm, 1, 32));
      rm = fmaxf(rm, __shfl_xor(rm, 2, 32));
      rm = fmaxf(rm, __shfl_xor(rm, 4, 32));
      rm = fmaxf(rm, __shfl_xor(rm, 8, 32));
      const float mn = fmaxf(m_run[r], rm);
      alpha[r] = __expf(m_run[r] - mn);
      const float p0 = __expf(x0 - mn);
      const float p1 = __expf(x1 - mn);
      float rs = p0 + p1;
      rs += __shfl_xor(rs, 1, 32);
      rs += __shfl_xor(rs, 2, 32);
      rs += __shfl_xor(rs, 4, 32);
      rs += __shfl_xor(rs, 8, 32);
      l_run[r] = l_run[r] * alpha[r] + rs;
      m_run[r] = mn;
      Ps[w][r + hi][n]      = (_Float16)p0;   // C layout -> LDS
      Ps[w][r + hi][16 + n] = (_Float16)p1;
    }
#pragma unroll
    for (int r = 0; r < 8; ++r) {
      o0[r] *= alpha[r]; o1[r] *= alpha[r]; o2[r] *= alpha[r]; o3[r] *= alpha[r];
    }

    // O += P(16x32 over keys) * V(32x64); V fragments from transposed tile
    const v16h ap = load_frag_a(&Ps[w][0][0], 40);
    o0 = wmma16(ap, load_frag_bt(&Vst[0][0],  40), o0);
    o1 = wmma16(ap, load_frag_bt(&Vst[16][0], 40), o1);
    o2 = wmma16(ap, load_frag_bt(&Vst[32][0], 40), o2);
    o3 = wmma16(ap, load_frag_bt(&Vst[48][0], 40), o3);
    __syncthreads();
  }

  // normalize and write [token][h*64 + d] as f16 for the output projection
#pragma unroll
  for (int r = 0; r < 8; ++r) {
    const float inv = 1.0f / l_run[r];
    const int tok = b * kSeq + q0 + hi + r;
    _Float16* dst = attn_out + (size_t)tok * kDim + h * kHd + n;
    dst[0]  = (_Float16)(o0[r] * inv);
    dst[16] = (_Float16)(o1[r] * inv);
    dst[32] = (_Float16)(o2[r] * inv);
    dst[48] = (_Float16)(o3[r] * inv);
  }
}

extern "C" void kernel_launch(void* const* d_in, const int* in_sizes, int n_in,
                              void* d_out, int out_size, void* d_ws, size_t ws_size,
                              hipStream_t stream) {
  (void)in_sizes; (void)n_in; (void)out_size; (void)ws_size;
  const float* x     = (const float*)d_in[0];
  const float* gamma = (const float*)d_in[1];
  const float* beta  = (const float*)d_in[2];
  const float* W_qkv = (const float*)d_in[3];
  const float* b_qkv = (const float*)d_in[4];
  const float* W_out = (const float*)d_in[5];
  const float* b_out = (const float*)d_in[6];

  char* ws = (char*)d_ws;
  size_t off = 0;
  auto take = [&](size_t bytes) {
    char* p = ws + off;
    off = (off + bytes + 255) & ~(size_t)255;
    return p;
  };
  _Float16* xn16    = (_Float16*)take((size_t)kTok * kDim * 2);   // 6.3 MB
  _Float16* wqkvT16 = (_Float16*)take((size_t)kDim * kQkv * 2);   // 3.5 MB [2304][768]
  _Float16* woutT16 = (_Float16*)take((size_t)kDim * kDim * 2);   // 1.2 MB [768][768]
  _Float16* qkv16   = (_Float16*)take((size_t)kTok * kQkv * 2);   // 18.9 MB
  _Float16* attn16  = (_Float16*)take((size_t)kTok * kDim * 2);   // 6.3 MB

  cvt_transpose_f16<<<dim3(kQkv / 32, kDim / 32), 256, 0, stream>>>(
      W_qkv, wqkvT16, kDim, kQkv);
  cvt_transpose_f16<<<dim3(kDim / 32, kDim / 32), 256, 0, stream>>>(
      W_out, woutT16, kDim, kDim);
  layernorm_to_f16<<<kTok, 256, 0, stream>>>(x, gamma, beta, xn16);
  gemm_wmma<true><<<dim3(kQkv / 128, kTok / 64), 256, 0, stream>>>(
      xn16, wqkvT16, b_qkv, qkv16, kTok, kQkv, kDim);
  attention_wmma<<<dim3(kSeq / 64, kBatch * kHeads), 128, 0, stream>>>(qkv16,
                                                                       attn16);
  gemm_wmma<false><<<dim3(kDim / 128, kTok / 64), 256, 0, stream>>>(
      attn16, woutT16, b_out, d_out, kTok, kDim, kDim);
}
